// Het_70274254897483
// MI455X (gfx1250) — matrix operand order repaired and checked by reading, not compiled, not run on hardware
//
#include <hip/hip_runtime.h>
#include <hip/hip_bf16.h>

typedef __bf16 bf16;
typedef unsigned int u32;
typedef __attribute__((ext_vector_type(16))) __bf16 v16bf;
typedef __attribute__((ext_vector_type(8)))  float  v8f;

#define BATCH 1024
#define NND   64
#define TSEQ  256
#define FSEQ  5
#define G3    192

// ---------------- WMMA fragment helpers (CDNA5 §7.12.2 layouts) -------------
// A 16x32 bf16: lane holds row m = lane&15; element e -> K = e + 8*((e>>3)+(lane>>4))
__device__ __forceinline__ v16bf load_a(const bf16* src, int ld) {
    int lane = threadIdx.x & 31;
    int m = lane & 15;
    v16bf a;
#pragma unroll
    for (int e = 0; e < 16; ++e) {
        int k = e + 8 * ((e >> 3) + (lane >> 4));
        a[e] = src[m * ld + k];
    }
    return a;
}
// B 32x16 from a row-major [N][K] buffer (implicit transpose): col n = lane&15,
// K = e + 8*((e>>3)+(lane>>4)). Each lane reads two 16-byte runs -> ds_load_b128 x2.
__device__ __forceinline__ v16bf load_bT(const bf16* src, int ld) {
    int lane = threadIdx.x & 31;
    int n = lane & 15;
    v16bf b;
#pragma unroll
    for (int e = 0; e < 16; ++e) b[e] = src[n * ld + e + 8 * ((e >> 3) + (lane >> 4))];
    return b;
}
__device__ __forceinline__ v8f wmma_bf16(v16bf a, v16bf b, v8f c) {
    return __builtin_amdgcn_wmma_f32_16x16x32_bf16(false, a, false, b, (short)0, c,
                                                   false, false);
}
__device__ __forceinline__ float leaky(float x) { return x >= 0.f ? x : 0.01f * x; }
__device__ __forceinline__ float sigmoidf(float x) { return 1.f / (1.f + __expf(-x)); }

// ---------------- fp32 -> bf16 conversion ----------------
__global__ void k_cvt(const float* __restrict__ src, bf16* __restrict__ dst, int n) {
    int i = blockIdx.x * 256 + threadIdx.x;
    if (i < n) dst[i] = (bf16)src[i];
}

// ---------------- normalized adjacency: Hn = D^-1/2 (A*(1-I)+I) D^-1/2 ------
__global__ void k_norm_adj(const float* __restrict__ A, bf16* __restrict__ Hn) {
    __shared__ float sH[64 * 64];
    __shared__ float dinv[64];
    int b = blockIdx.x, tid = threadIdx.x;
    const float* Ab = A + (size_t)b * 4096;
    for (int i = tid; i < 4096; i += 256) {
        int r = i >> 6, c = i & 63;
        sH[i] = (r == c) ? 1.0f : Ab[i];
    }
    __syncthreads();
    if (tid < 64) {
        float s = 0.f;
        for (int j = 0; j < 64; ++j) s += sH[tid * 64 + j];
        dinv[tid] = rsqrtf(s);
    }
    __syncthreads();
    bf16* out = Hn + (size_t)b * 4096;
    for (int i = tid; i < 4096; i += 256) {
        int r = i >> 6, c = i & 63;
        out[i] = (bf16)(dinv[r] * sH[i] * dinv[c]);
    }
}

// ---------------- GCN: Xp[b][s][n] = leaky(w * (Hn @ X)[n][s]) --------------
// grid (B, ceil(S/64)), block 128. B tile staged transposed [c][k] so fragment
// loads are contiguous (ds_load_b128).
__global__ void k_gcn(const float* __restrict__ X, const bf16* __restrict__ Hn,
                      const float* __restrict__ wscal, bf16* __restrict__ Xp, int S) {
    __shared__ bf16 sA[64 * 64];   // Hn (A operand, row-major [m][k])
    __shared__ bf16 sBt[64 * 64];  // X tile, [c][k]-major
    int b = blockIdx.x, c0 = blockIdx.y * 64, tid = threadIdx.x;
    int wave = tid >> 5, lane = tid & 31;
    const bf16* Hb = Hn + (size_t)b * 4096;
    for (int i = tid; i < 2048; i += 128) ((u32*)sA)[i] = ((const u32*)Hb)[i];
    for (int i = tid; i < 4096; i += 128) {
        int k = i >> 6, c = i & 63;  // consecutive tid -> consecutive c (coalesced)
        int s = c0 + c;
        sBt[c * 64 + k] =
            (s < S) ? (bf16)X[(size_t)b * 64 * S + (size_t)k * S + s] : (bf16)0.f;
    }
    __syncthreads();
    float w = wscal[0];
    int n0 = wave * 16;
    v16bf b0 = load_bT(sBt + n0 * 64, 64);       // K 0..31
    v16bf b1 = load_bT(sBt + n0 * 64 + 32, 64);  // K 32..63
    v8f acc[4] = {};
#pragma unroll
    for (int mt = 0; mt < 4; ++mt) {
        v16bf a0 = load_a(sA + mt * 16 * 64, 64);
        v16bf a1 = load_a(sA + mt * 16 * 64 + 32, 64);
        acc[mt] = wmma_bf16(a0, b0, acc[mt]);
        acc[mt] = wmma_bf16(a1, b1, acc[mt]);
    }
    int col = lane & 15, mh = 8 * (lane >> 4);
    int s = c0 + n0 + col;
#pragma unroll
    for (int mt = 0; mt < 4; ++mt)
#pragma unroll
        for (int v = 0; v < 8; ++v) {
            int m = mt * 16 + mh + v;
            if (s < S)
                Xp[((size_t)b * S + s) * 64 + m] = (bf16)leaky(w * acc[mt][v]);
        }
}

// ---------------- persistent bi-GRU step kernel -----------------------------
// One block = 64 batch rows, one direction. h kept in LDS; per-step
// x_s@Wih^T and h@Whh^T are 64x192x64 WMMA GEMMs (weights bf16, cache-resident).
__global__ void k_gru(const bf16* __restrict__ Xp, const bf16* __restrict__ wih,
                      const bf16* __restrict__ whh, const float* __restrict__ bih,
                      const float* __restrict__ bhh, bf16* __restrict__ o, int S,
                      int dir) {
    __shared__ bf16 sx[64 * 64];
    __shared__ bf16 sh[64 * 64];
    __shared__ float shf[64 * 64];
    int b0 = blockIdx.x * 64, tid = threadIdx.x, wave = tid >> 5, lane = tid & 31;
    for (int i = tid; i < 4096; i += 128) {
        sh[i] = (bf16)0.f;
        shf[i] = 0.f;
    }
    __syncthreads();
    int m0 = wave * 16;
    int col = lane & 15, mh = 8 * (lane >> 4);
    for (int ss = 0; ss < S; ++ss) {
        int s = dir ? (S - 1 - ss) : ss;
        // prefetch next step's x tile while this step computes
        if (ss + 1 < S) {
            int sn = dir ? (S - 2 - ss) : (ss + 1);
            __builtin_prefetch(
                (const void*)(Xp + ((size_t)(b0 + (tid >> 1)) * S + sn) * 64), 0, 0);
        }
        for (int i = tid; i < 2048; i += 128) {  // stage x_s tile (bf16, u32 copies)
            int r = i >> 5, p = i & 31;
            ((u32*)sx)[r * 32 + p] =
                ((const u32*)(Xp + ((size_t)(b0 + r) * S + s) * 64))[p];
        }
        __syncthreads();
        v16bf ax0 = load_a(sx + m0 * 64, 64);
        v16bf ax1 = load_a(sx + m0 * 64 + 32, 64);
        v16bf ah0 = load_a(sh + m0 * 64, 64);
        v16bf ah1 = load_a(sh + m0 * 64 + 32, 64);
        v8f gx[12] = {};
        v8f gh[12] = {};
#pragma unroll
        for (int j = 0; j < 12; ++j) {
            v16bf bx0 = load_bT(wih + j * 16 * 64, 64);
            v16bf bx1 = load_bT(wih + j * 16 * 64 + 32, 64);
            gx[j] = wmma_bf16(ax0, bx0, gx[j]);
            gx[j] = wmma_bf16(ax1, bx1, gx[j]);
            v16bf bh0 = load_bT(whh + j * 16 * 64, 64);
            v16bf bh1 = load_bT(whh + j * 16 * 64 + 32, 64);
            gh[j] = wmma_bf16(ah0, bh0, gh[j]);
            gh[j] = wmma_bf16(ah1, bh1, gh[j]);
        }
        __syncthreads();  // all reads of sh done before updating h
#pragma unroll
        for (int j = 0; j < 4; ++j) {
            int hh = j * 16 + col;
            float bxr = bih[hh], bhr = bhh[hh];
            float bxz = bih[64 + hh], bhz = bhh[64 + hh];
            float bxn = bih[128 + hh], bhn = bhh[128 + hh];
#pragma unroll
            for (int v = 0; v < 8; ++v) {
                int m = m0 + mh + v;
                float r = sigmoidf(gx[j][v] + bxr + gh[j][v] + bhr);
                float z = sigmoidf(gx[4 + j][v] + bxz + gh[4 + j][v] + bhz);
                float n = tanhf(gx[8 + j][v] + bxn + r * (gh[8 + j][v] + bhn));
                float hn2 = (1.f - z) * n + z * shf[m * 64 + hh];
                shf[m * 64 + hh] = hn2;
                sh[m * 64 + hh] = (bf16)hn2;
                o[((size_t)(b0 + m) * S + s) * 128 + dir * 64 + hh] = (bf16)hn2;
            }
        }
        __syncthreads();
    }
}

// ---------------- batchnorm statistics (training-mode batch stats) ----------
__global__ void k_bnstats(const bf16* __restrict__ o, const float* __restrict__ g,
                          const float* __restrict__ beta, float* __restrict__ scale,
                          float* __restrict__ shift, int S) {
    __shared__ float rs[256], rq[256];
    int s = blockIdx.x, tid = threadIdx.x;
    float sum = 0.f, sq = 0.f;
    for (int i = tid; i < BATCH * 128; i += 256) {
        int b = i >> 7, c = i & 127;
        float x = (float)o[((size_t)b * S + s) * 128 + c];
        sum += x;
        sq += x * x;
    }
    rs[tid] = sum;
    rq[tid] = sq;
    __syncthreads();
    for (int st = 128; st > 0; st >>= 1) {
        if (tid < st) {
            rs[tid] += rs[tid + st];
            rq[tid] += rq[tid + st];
        }
        __syncthreads();
    }
    if (tid == 0) {
        float cnt = (float)(BATCH * 128);
        float m = rs[0] / cnt;
        float var = rq[0] / cnt - m * m;
        float sc = g[s] * rsqrtf(var + 1e-5f);
        scale[s] = sc;
        shift[s] = beta[s] - m * sc;
    }
}

// ---------------- fold BN into head weights: W'[t2][k]=W[t2][k]*scale[k] ----
__global__ void k_headprep(const float* __restrict__ lw, const float* __restrict__ lb,
                           const float* __restrict__ scale,
                           const float* __restrict__ shift, bf16* __restrict__ Wp,
                           float* __restrict__ bp, int S, int Kpad) {
    int t2 = threadIdx.x;  // 128 threads
    float acc = lb[t2];
    for (int s2 = 0; s2 < S; ++s2) acc += shift[s2] * lw[t2 * S + s2];
    bp[t2] = acc;
    for (int k = 0; k < Kpad; ++k)
        Wp[t2 * Kpad + k] = (k < S) ? (bf16)(lw[t2 * S + k] * scale[k]) : (bf16)0.f;
}

// ---------------- batched head GEMM: out[b][c*128+t2] = sum_s o*W' + b' -----
__global__ void k_head(const bf16* __restrict__ o, const bf16* __restrict__ Wp,
                       const float* __restrict__ bp, bf16* __restrict__ out, int S,
                       int Kpad, int off) {
    __shared__ bf16 sA[128 * 32];   // W' slab [t2][k]
    __shared__ bf16 sBt[128 * 32];  // o slab, [c][k]-major
    int b = blockIdx.x, tid = threadIdx.x, wave = tid >> 5, lane = tid & 31;
    v8f acc[16] = {};
    for (int kk = 0; kk < Kpad; kk += 32) {
        __syncthreads();
        for (int i = tid; i < 2048; i += 128) {
            int r = i >> 4, p = i & 15;
            ((u32*)sA)[i] = ((const u32*)(Wp + r * Kpad + kk))[p];
        }
        // stage o transposed: read u32 (2 channels) coalesced, scatter into [c][k]
        for (int i = tid; i < 2048; i += 128) {
            int k = i >> 6, c2 = i & 63;  // consecutive tid -> consecutive c pair
            int s = kk + k;
            u32 v = (s < S) ? ((const u32*)(o + ((size_t)b * S + s) * 128))[c2] : 0u;
            bf16 lo = (bf16)0.f, hi = (bf16)0.f;
            *(u32*)&lo == 0u;  // no-op to keep types simple
            union { u32 u; bf16 h[2]; } cv;
            cv.u = v;
            sBt[(2 * c2 + 0) * 32 + k] = cv.h[0];
            sBt[(2 * c2 + 1) * 32 + k] = cv.h[1];
            (void)lo; (void)hi;
        }
        __syncthreads();
#pragma unroll
        for (int mt = 0; mt < 2; ++mt) {
            v16bf a = load_a(sA + (wave * 32 + mt * 16) * 32, 32);
#pragma unroll
            for (int nt = 0; nt < 8; ++nt) {
                v16bf bb = load_bT(sBt + nt * 16 * 32, 32);
                acc[mt * 8 + nt] = wmma_bf16(a, bb, acc[mt * 8 + nt]);
            }
        }
    }
    int col = lane & 15, mh = 8 * (lane >> 4);
#pragma unroll
    for (int mt = 0; mt < 2; ++mt)
#pragma unroll
        for (int nt = 0; nt < 8; ++nt)
#pragma unroll
            for (int v = 0; v < 8; ++v) {
                int t2 = wave * 32 + mt * 16 + mh + v;
                int c = nt * 16 + col;
                out[(size_t)b * 32768 + off + c * 128 + t2] =
                    (bf16)(acc[mt * 8 + nt][v] + bp[t2]);
            }
}

// ---------------- lin1 (1024x64x32768 WMMA) + fused leaky + lin2 ------------
__global__ void k_lin(const bf16* __restrict__ cat, const float* __restrict__ w1,
                      const float* __restrict__ b1, const float* __restrict__ w2,
                      const float* __restrict__ b2, float* __restrict__ out) {
    __shared__ bf16 sA[64 * 32];   // activations [m][k]
    __shared__ bf16 sBt[64 * 32];  // w1 slab [j][k]
    __shared__ float sh1[64 * 64];
    int b0 = blockIdx.x * 64, tid = threadIdx.x, wave = tid >> 5, lane = tid & 31;
    v8f acc[4] = {};
    for (int kk = 0; kk < 32768; kk += 32) {
        // prefetch next K slab (streaming 64MB bf16 cat: the bandwidth-bound loop)
        if (kk + 32 < 32768) {
            __builtin_prefetch(
                (const void*)(cat + (size_t)(b0 + (tid >> 1)) * 32768 + kk + 32), 0, 0);
            __builtin_prefetch(
                (const void*)(w1 + (size_t)(tid >> 1) * 32768 + kk + 32), 0, 0);
        }
        __syncthreads();
        for (int i = tid; i < 1024; i += 128) {
            int r = i >> 4, p = i & 15;
            ((u32*)sA)[i] = ((const u32*)(cat + (size_t)(b0 + r) * 32768 + kk))[p];
        }
        for (int i = tid; i < 2048; i += 128) {
            int j = i >> 5, k = i & 31;
            sBt[j * 32 + k] = (bf16)w1[(size_t)j * 32768 + kk + k];
        }
        __syncthreads();
        v16bf a = load_a(sA + wave * 16 * 32, 32);
#pragma unroll
        for (int nt = 0; nt < 4; ++nt) {
            v16bf bb = load_bT(sBt + nt * 16 * 32, 32);
            acc[nt] = wmma_bf16(a, bb, acc[nt]);
        }
    }
    int col = lane & 15, mh = 8 * (lane >> 4);
#pragma unroll
    for (int nt = 0; nt < 4; ++nt)
#pragma unroll
        for (int v = 0; v < 8; ++v) {
            int m = wave * 16 + mh + v, j = nt * 16 + col;
            sh1[m * 64 + j] = leaky(acc[nt][v] + b1[j]);
        }
    __syncthreads();
    for (int idx = tid; idx < 64 * 3; idx += 128) {
        int m = idx / 3, i = idx - 3 * m;
        float d = b2[i];
        for (int j = 0; j < 64; ++j) d += sh1[m * 64 + j] * w2[i * 64 + j];
        out[(size_t)(b0 + m) * 3 + i] = d;
    }
}

// ---------------- host side -------------------------------------------------
extern "C" void kernel_launch(void* const* d_in, const int* in_sizes, int n_in,
                              void* d_out, int out_size, void* d_ws, size_t ws_size,
                              hipStream_t stream) {
    (void)in_sizes; (void)n_in; (void)out_size; (void)ws_size;
    const float* x_T   = (const float*)d_in[0];
    const float* x_F   = (const float*)d_in[1];
    const float* A     = (const float*)d_in[2];
    const float* gcnwF = (const float*)d_in[3];
    const float* gcnwT = (const float*)d_in[14];
    const float* grF[8] = {(const float*)d_in[4],  (const float*)d_in[5],
                           (const float*)d_in[6],  (const float*)d_in[7],
                           (const float*)d_in[8],  (const float*)d_in[9],
                           (const float*)d_in[10], (const float*)d_in[11]};
    const float* grT[8] = {(const float*)d_in[15], (const float*)d_in[16],
                           (const float*)d_in[17], (const float*)d_in[18],
                           (const float*)d_in[19], (const float*)d_in[20],
                           (const float*)d_in[21], (const float*)d_in[22]};
    const float* bnFg = (const float*)d_in[12];
    const float* bnFb = (const float*)d_in[13];
    const float* bnTg = (const float*)d_in[23];
    const float* bnTb = (const float*)d_in[24];
    const float* linFw = (const float*)d_in[25];
    const float* linFb = (const float*)d_in[26];
    const float* linTw = (const float*)d_in[27];
    const float* linTb = (const float*)d_in[28];
    const float* w1 = (const float*)d_in[29];
    const float* b1 = (const float*)d_in[30];
    const float* w2 = (const float*)d_in[31];
    const float* b2 = (const float*)d_in[32];
    float* out = (float*)d_out;

    char* ws = (char*)d_ws;
    size_t off = 0;
    auto alloc = [&](size_t bytes) -> char* {
        char* p = ws + off;
        off += (bytes + 255) & ~(size_t)255;
        return p;
    };
    bf16* d_Hn  = (bf16*)alloc((size_t)BATCH * 4096 * 2);
    bf16* d_XpT = (bf16*)alloc((size_t)BATCH * TSEQ * 64 * 2);
    bf16* d_XpF = (bf16*)alloc((size_t)BATCH * FSEQ * 64 * 2);
    bf16* d_oT  = (bf16*)alloc((size_t)BATCH * TSEQ * 128 * 2);
    bf16* d_oF  = (bf16*)alloc((size_t)BATCH * FSEQ * 128 * 2);
    bf16* d_cat = (bf16*)alloc((size_t)BATCH * 32768 * 2);
    bf16* wTf_ih = (bf16*)alloc(G3 * 64 * 2);
    bf16* wTf_hh = (bf16*)alloc(G3 * 64 * 2);
    bf16* wTb_ih = (bf16*)alloc(G3 * 64 * 2);
    bf16* wTb_hh = (bf16*)alloc(G3 * 64 * 2);
    bf16* wFf_ih = (bf16*)alloc(G3 * 64 * 2);
    bf16* wFf_hh = (bf16*)alloc(G3 * 64 * 2);
    bf16* wFb_ih = (bf16*)alloc(G3 * 64 * 2);
    bf16* wFb_hh = (bf16*)alloc(G3 * 64 * 2);
    float* scT = (float*)alloc(TSEQ * 4);
    float* shT = (float*)alloc(TSEQ * 4);
    float* scF = (float*)alloc(64);
    float* shF = (float*)alloc(64);
    bf16*  WpT = (bf16*)alloc(128 * 256 * 2);
    float* bpT = (float*)alloc(128 * 4);
    bf16*  WpF = (bf16*)alloc(128 * 32 * 2);
    float* bpF = (float*)alloc(128 * 4);

    // weight conversions to bf16
    const int nw = G3 * 64, gw = (nw + 255) / 256;
    k_cvt<<<gw, 256, 0, stream>>>(grT[0], wTf_ih, nw);
    k_cvt<<<gw, 256, 0, stream>>>(grT[1], wTf_hh, nw);
    k_cvt<<<gw, 256, 0, stream>>>(grT[4], wTb_ih, nw);
    k_cvt<<<gw, 256, 0, stream>>>(grT[5], wTb_hh, nw);
    k_cvt<<<gw, 256, 0, stream>>>(grF[0], wFf_ih, nw);
    k_cvt<<<gw, 256, 0, stream>>>(grF[1], wFf_hh, nw);
    k_cvt<<<gw, 256, 0, stream>>>(grF[4], wFb_ih, nw);
    k_cvt<<<gw, 256, 0, stream>>>(grF[5], wFb_hh, nw);

    k_norm_adj<<<BATCH, 256, 0, stream>>>(A, d_Hn);

    k_gcn<<<dim3(BATCH, 4), 128, 0, stream>>>(x_T, d_Hn, gcnwT, d_XpT, TSEQ);
    k_gcn<<<dim3(BATCH, 1), 128, 0, stream>>>(x_F, d_Hn, gcnwF, d_XpF, FSEQ);

    k_gru<<<BATCH / 64, 128, 0, stream>>>(d_XpT, wTf_ih, wTf_hh, grT[2], grT[3],
                                          d_oT, TSEQ, 0);
    k_gru<<<BATCH / 64, 128, 0, stream>>>(d_XpT, wTb_ih, wTb_hh, grT[6], grT[7],
                                          d_oT, TSEQ, 1);
    k_gru<<<BATCH / 64, 128, 0, stream>>>(d_XpF, wFf_ih, wFf_hh, grF[2], grF[3],
                                          d_oF, FSEQ, 0);
    k_gru<<<BATCH / 64, 128, 0, stream>>>(d_XpF, wFb_ih, wFb_hh, grF[6], grF[7],
                                          d_oF, FSEQ, 1);

    k_bnstats<<<TSEQ, 256, 0, stream>>>(d_oT, bnTg, bnTb, scT, shT, TSEQ);
    k_bnstats<<<FSEQ, 256, 0, stream>>>(d_oF, bnFg, bnFb, scF, shF, FSEQ);

    k_headprep<<<1, 128, 0, stream>>>(linTw, linTb, scT, shT, WpT, bpT, TSEQ, 256);
    k_headprep<<<1, 128, 0, stream>>>(linFw, linFb, scF, shF, WpF, bpF, FSEQ, 32);

    k_head<<<BATCH, 128, 0, stream>>>(d_oF, WpF, bpF, d_cat, FSEQ, 32, 0);
    k_head<<<BATCH, 128, 0, stream>>>(d_oT, WpT, bpT, d_cat, TSEQ, 256, 16384);

    k_lin<<<BATCH / 64, 128, 0, stream>>>(d_cat, w1, b1, w2, b2, out);
}
